// NodeEmbedding_54193897341343
// MI455X (gfx1250) — matrix operand order repaired and checked by reading, not compiled, not run on hardware
//
#include <hip/hip_runtime.h>
#include <stdint.h>

// ---------------- CDNA5 vector types ----------------
typedef __attribute__((ext_vector_type(16))) __bf16   v16bf;
typedef __attribute__((ext_vector_type(8)))  float    v8f;
typedef __attribute__((ext_vector_type(4)))  uint32_t u32x4;

// ---------------- problem constants -----------------
// dims: 203 -> 175 -> 150 -> 128 -> 80 -> 48
static constexpr int FANIN [5] = {203, 175, 150, 128, 80};
static constexpr int FANOUT[5] = {175, 150, 128, 80, 48};
static constexpr int KP[5] = {224, 192, 160, 128, 96};  // fan_in  padded to mult of 32 (WMMA K)
static constexpr int NP[5] = {176, 160, 128, 80, 48};   // fan_out padded to mult of 16 (WMMA N)
static constexpr int WOFF[5] = {0, 39424, 70144, 90624, 100864}; // prefix sums of NP*KP
static constexpr int WTOT = 105472;                              // total padded weight elems
static constexpr int BOFF[5] = {0, 176, 336, 464, 544};          // prefix sums of NP
static constexpr int BTOT = 592;

// ---------------- bf16 split helpers ----------------
__device__ __forceinline__ uint16_t f2bf(float f) {   // round-to-nearest-even
  uint32_t u = __builtin_bit_cast(uint32_t, f);
  u += 0x7FFFu + ((u >> 16) & 1u);
  return (uint16_t)(u >> 16);
}
__device__ __forceinline__ float bf2f(uint16_t h) {
  return __builtin_bit_cast(float, ((uint32_t)h) << 16);
}

union V16 { u32x4 q[2]; v16bf v; };

// A fragment (16x32 bf16): per-lane K runs [base, base+8) and [base+16, base+24)
__device__ __forceinline__ v16bf loadA(const uint16_t* p) {
  V16 t;
  t.q[0] = *(const u32x4*)(p);
  t.q[1] = *(const u32x4*)(p + 16);
  return t.v;
}
// B fragment (32x16 bf16): per-lane K runs [base, base+8) and [base+16, base+24)
__device__ __forceinline__ v16bf loadB(const uint16_t* p) {
  V16 t;
  t.q[0] = *(const u32x4*)(p);
  t.q[1] = *(const u32x4*)(p + 16);
  return t.v;
}

__device__ __forceinline__ v8f wmma_bf16(v16bf a, v16bf b, v8f c) {
  // (neg_a, A, neg_b, B, c_mod, C, reuse_a, reuse_b)
  return __builtin_amdgcn_wmma_f32_16x16x32_bf16(false, a, false, b, (short)0, c,
                                                 false, false);
}

__device__ __forceinline__ float fast_tanh(float x) {
  // 1 - 2/(exp(2x)+1): exact tanh identity; saturates correctly for |x| large
  return 1.0f - 2.0f / (__expf(2.0f * x) + 1.0f);
}

// ---------------- one fused layer -------------------
// Input activations (bf16 hi/lo planes, row-major [16][KPIN]) live in the wave's
// private LDS buffer; A fragments are hoisted into registers first, so the same
// buffer is safely overwritten with the layer's output ([16][KPOUT]).
template <int KT, int NT, int KPIN, int KPOUT, bool ACT, bool LAST>
__device__ __forceinline__ void run_layer(uint16_t* actHi, uint16_t* actLo,
                                          const uint16_t* __restrict__ wHi,
                                          const uint16_t* __restrict__ wLo,
                                          const float* __restrict__ bias,
                                          float* __restrict__ gout, int lane) {
  const int m    = lane & 15;
  const int aoff = (lane < 16) ? 0 : 8;    // A-fragment K sub-offset (also C row offset)
  const int boff = (lane < 16) ? 0 : 8;    // B-fragment K sub-offset

  // 1) whole input row-block -> registers (hi & lo planes)
  v16bf ahi[KT], alo[KT];
#pragma unroll
  for (int kt = 0; kt < KT; ++kt) {
    const int base = m * KPIN + kt * 32 + aoff;
    ahi[kt] = loadA(actHi + base);
    alo[kt] = loadA(actLo + base);
  }

  // 2) zero-fill K-padding columns of the output buffer (tanh(0)=0 handled by
  //    zero-padded weights/bias for cols < NT*16; cols in [NT*16, KPOUT) here)
  if constexpr (!LAST && (KPOUT > NT * 16)) {
    constexpr int TAILW = KPOUT - NT * 16;
    for (int i = lane; i < 16 * TAILW; i += 32) {
      const int r = i / TAILW, c = NT * 16 + (i % TAILW);
      actHi[r * KPOUT + c] = 0;
      actLo[r * KPOUT + c] = 0;
    }
  }

  // 3) N-tile loop: stream B fragments from L2, 3 WMMAs per K-tile (split-bf16)
  for (int nt = 0; nt < NT; ++nt) {
    const int wrow = (nt * 16 + m) * KPIN + boff;
    if (nt + 1 < NT) {  // hint next weight rows into cache
      __builtin_prefetch(wHi + wrow + 16 * KPIN, 0, 0);
      __builtin_prefetch(wLo + wrow + 16 * KPIN, 0, 0);
    }
    v8f acc = {};
#pragma unroll
    for (int kt = 0; kt < KT; ++kt) {
      const v16bf bhi = loadB(wHi + wrow + kt * 32);
      const v16bf blo = loadB(wLo + wrow + kt * 32);
      acc = wmma_bf16(ahi[kt], bhi, acc);  // hi*hi
      acc = wmma_bf16(ahi[kt], blo, acc);  // hi*lo
      acc = wmma_bf16(alo[kt], bhi, acc);  // lo*hi
    }
    // C/D layout: VGPR r of a lane -> (M = r + aoff, N = m); one bias per lane
    const float bv = bias[nt * 16 + m];
#pragma unroll
    for (int r = 0; r < 8; ++r) {
      float v = acc[r] + bv;
      if constexpr (ACT) v = fast_tanh(v);
      const int row = r + aoff;
      const int col = nt * 16 + m;
      if constexpr (LAST) {
        gout[row * 48 + col] = v;
      } else {
        const uint16_t h = f2bf(v);
        actHi[row * KPOUT + col] = h;
        actLo[row * KPOUT + col] = f2bf(v - bf2f(h));
      }
    }
  }
}

// ---------------- main fused kernel -----------------
#define WAVES 4  // 4 waves/block, 16 rows per wave -> 64 batch rows per block

__global__ __launch_bounds__(32 * WAVES) void mlp_fused_kernel(
    const float* __restrict__ x, const uint16_t* __restrict__ wHi,
    const uint16_t* __restrict__ wLo, const float* __restrict__ bias,
    float* __restrict__ out) {
  __shared__ __align__(16) uint16_t sHi[WAVES][16 * 224];  // 28,672 B
  __shared__ __align__(16) uint16_t sLo[WAVES][16 * 224];  // 28,672 B
  const int lane = threadIdx.x & 31;
  const int wave = threadIdx.x >> 5;
  uint16_t* hi = &sHi[wave][0];
  uint16_t* lo = &sLo[wave][0];
  const long rowBase = (long)blockIdx.x * (16 * WAVES) + (long)wave * 16;

  // stage this wave's 16 input rows: fp32 -> split bf16, zero-padded to K=224
  for (int r = 0; r < 16; ++r) {
    const float* xr = x + (rowBase + r) * 203;
    for (int c = lane; c < 224; c += 32) {
      const float v = (c < 203) ? xr[c] : 0.0f;
      const uint16_t h = f2bf(v);
      hi[r * 224 + c] = h;
      lo[r * 224 + c] = f2bf(v - bf2f(h));
    }
  }

  //            KT  NT  KPIN KPOUT  ACT    LAST
  run_layer<7, 11, 224, 192, true,  false>(hi, lo, wHi + WOFF[0], wLo + WOFF[0], bias + BOFF[0], nullptr, lane);
  run_layer<6, 10, 192, 160, true,  false>(hi, lo, wHi + WOFF[1], wLo + WOFF[1], bias + BOFF[1], nullptr, lane);
  run_layer<5,  8, 160, 128, true,  false>(hi, lo, wHi + WOFF[2], wLo + WOFF[2], bias + BOFF[2], nullptr, lane);
  run_layer<4,  5, 128,  96, true,  false>(hi, lo, wHi + WOFF[3], wLo + WOFF[3], bias + BOFF[3], nullptr, lane);
  run_layer<3,  3,  96,   0, false, true >(hi, lo, wHi + WOFF[4], wLo + WOFF[4], bias + BOFF[4],
                                           out + rowBase * 48, lane);
}

// ---------------- weight prep kernel ----------------
// fp32 W[fan_out][fan_in] -> zero-padded bf16 hi/lo planes [NP][KP] in d_ws,
// plus zero-padded fp32 bias. Layout lets B-fragment loads be contiguous 16 B runs.
__global__ void prep_kernel(const float* __restrict__ W1, const float* __restrict__ b1,
                            const float* __restrict__ W2, const float* __restrict__ b2,
                            const float* __restrict__ W3, const float* __restrict__ b3,
                            const float* __restrict__ W4, const float* __restrict__ b4,
                            const float* __restrict__ W5, const float* __restrict__ b5,
                            uint16_t* __restrict__ wHi, uint16_t* __restrict__ wLo,
                            float* __restrict__ biasOut) {
  const float* W[5]  = {W1, W2, W3, W4, W5};
  const float* bb[5] = {b1, b2, b3, b4, b5};
  const int idx = blockIdx.x * blockDim.x + threadIdx.x;
  if (idx < WTOT) {
    int l = 0;
    while (l < 4 && idx >= WOFF[l + 1]) ++l;
    const int rel = idx - WOFF[l];
    const int n = rel / KP[l], k = rel % KP[l];
    const float v = (n < FANOUT[l] && k < FANIN[l]) ? W[l][n * FANIN[l] + k] : 0.0f;
    const uint16_t h = f2bf(v);
    wHi[idx] = h;
    wLo[idx] = f2bf(v - bf2f(h));
  } else if (idx < WTOT + BTOT) {
    const int r = idx - WTOT;
    int l = 0;
    while (l < 4 && r >= BOFF[l + 1]) ++l;
    const int n = r - BOFF[l];
    biasOut[r] = (n < FANOUT[l]) ? bb[l][n] : 0.0f;
  }
}

// ---------------- launch ----------------------------
extern "C" void kernel_launch(void* const* d_in, const int* in_sizes, int n_in,
                              void* d_out, int out_size, void* d_ws, size_t ws_size,
                              hipStream_t stream) {
  (void)in_sizes; (void)n_in; (void)out_size; (void)ws_size;
  const float* x  = (const float*)d_in[0];
  const float* W1 = (const float*)d_in[1];  const float* b1 = (const float*)d_in[2];
  const float* W2 = (const float*)d_in[3];  const float* b2 = (const float*)d_in[4];
  const float* W3 = (const float*)d_in[5];  const float* b3 = (const float*)d_in[6];
  const float* W4 = (const float*)d_in[7];  const float* b4 = (const float*)d_in[8];
  const float* W5 = (const float*)d_in[9];  const float* b5 = (const float*)d_in[10];

  // workspace layout: [hi plane | lo plane | bias]  (~424 KB total)
  uint16_t* wHi = (uint16_t*)d_ws;
  uint16_t* wLo = wHi + WTOT;
  float*    bias = (float*)(wHi + 2 * WTOT);

  const int prepThreads = WTOT + BTOT;
  prep_kernel<<<(prepThreads + 255) / 256, 256, 0, stream>>>(
      W1, b1, W2, b2, W3, b3, W4, b4, W5, b5, wHi, wLo, bias);

  const int B = 131072;
  mlp_fused_kernel<<<B / (16 * WAVES), 32 * WAVES, 0, stream>>>(
      x, wHi, wLo, bias, (float*)d_out);
}
